// ReverseGRNLayer_71167608095135
// MI455X (gfx1250) — matrix operand order repaired and checked by reading, not compiled, not run on hardware
//
#include <hip/hip_runtime.h>
#include <hip/hip_bf16.h>

typedef __attribute__((ext_vector_type(16))) _Float16 v16h;
typedef __attribute__((ext_vector_type(8)))  float    v8f;
typedef __attribute__((ext_vector_type(4)))  float    f32x4;

union Frag16 { v16h h; f32x4 f[2]; };

// ---------------------------------------------------------------------------
// Fragment loads for V_WMMA_F32_16X16X32_F16 (wave32), operands K-major.
// A (16x32 MxK): lanes 0-15 row M=lane, K 0..7 (V0-3) & 16..23 (V4-7);
//                lanes 16-31 row M=lane-16, K 8..15 & 24..31.
// B (32x16 KxN): lane n<16 col n K 0..7 & 16..23; lane n+16 col n K 8..15 & 24..31,
//                two 16-bit values per VGPR, K consecutive.
// ---------------------------------------------------------------------------
static __device__ __forceinline__ v16h load_fragA(const _Float16* __restrict__ base,
                                                  int row_stride, int lane) {
  int r  = lane & 15;
  int kh = (lane >> 4) * 8;
  const _Float16* p = base + (long)r * row_stride + kh;
  Frag16 fr;
  fr.f[0] = *(const f32x4*)(p);
  fr.f[1] = *(const f32x4*)(p + 16);
  return fr.h;
}

static __device__ __forceinline__ v16h load_fragB(const _Float16* __restrict__ base,
                                                  int row_stride, int lane) {
  int c  = lane & 15;
  int kh = (lane >> 4) * 8;
  const _Float16* p = base + (long)c * row_stride + kh;
  Frag16 fr;
  fr.f[0] = *(const f32x4*)(p);
  fr.f[1] = *(const f32x4*)(p + 16);
  return fr.h;
}

static __device__ __forceinline__ v8f wmma_f16(v16h a, v16h b, v8f c) {
  return __builtin_amdgcn_wmma_f32_16x16x32_f16(false, a, false, b, (short)0, c,
                                                false, false);
}

static __device__ __forceinline__ float sigmoidf(float x) {
  return 1.0f / (1.0f + __expf(-x));
}

// ---------------------------------------------------------------------------
// Split f32 -> (f16 hi, f16 lo) so hi+lo reproduces ~22 mantissa bits.
// ---------------------------------------------------------------------------
__global__ void split_f32_kernel(const float* __restrict__ x,
                                 _Float16* __restrict__ hi,
                                 _Float16* __restrict__ lo, int n) {
  int i = blockIdx.x * blockDim.x + threadIdx.x;
  int stride = gridDim.x * blockDim.x;
  for (; i < n; i += stride) {
    float v = x[i];
    _Float16 h = (_Float16)v;
    hi[i] = h;
    lo[i] = (_Float16)(v - (float)h);
  }
}

// ---------------------------------------------------------------------------
// Phase 1: Xp[M=65536, N=2048] = A[M,512] * W_ih[N,512]^T + (b_ih + b_hh)
// Block tile 128x128, 8 waves, wave tile 64(M)x32(N).
// Split-f16: acc += Ah*Bh + Al*Bh + Ah*Bl  (3 WMMAs per tile-kstep)
// ---------------------------------------------------------------------------
__global__ __launch_bounds__(256) void xproj_gemm_kernel(
    const _Float16* __restrict__ Ah, const _Float16* __restrict__ Al,
    const _Float16* __restrict__ Bh, const _Float16* __restrict__ Bl,
    const float* __restrict__ b_ih, const float* __restrict__ b_hh,
    float* __restrict__ Xp) {
  const int K = 512, N = 2048;
  int lane  = threadIdx.x & 31;
  int wave  = threadIdx.x >> 5;
  int mwave = wave & 1;      // 2 wave rows
  int nwave = wave >> 1;     // 4 wave cols
  long Mbase = (long)blockIdx.y * 128 + mwave * 64;
  int  Nbase = blockIdx.x * 128 + nwave * 32;

  v8f acc[4][2] = {};

  for (int kb = 0; kb < K; kb += 32) {
    v16h afh[4], afl[4];
#pragma unroll
    for (int mt = 0; mt < 4; mt++) {
      afh[mt] = load_fragA(Ah + (Mbase + mt * 16) * (long)K + kb, K, lane);
      afl[mt] = load_fragA(Al + (Mbase + mt * 16) * (long)K + kb, K, lane);
    }
    v16h bfh[2], bfl[2];
#pragma unroll
    for (int nt = 0; nt < 2; nt++) {
      bfh[nt] = load_fragB(Bh + (long)(Nbase + nt * 16) * K + kb, K, lane);
      bfl[nt] = load_fragB(Bl + (long)(Nbase + nt * 16) * K + kb, K, lane);
    }
#pragma unroll
    for (int mt = 0; mt < 4; mt++)
#pragma unroll
      for (int nt = 0; nt < 2; nt++) {
        acc[mt][nt] = wmma_f16(afh[mt], bfh[nt], acc[mt][nt]);
        acc[mt][nt] = wmma_f16(afl[mt], bfh[nt], acc[mt][nt]);
        acc[mt][nt] = wmma_f16(afh[mt], bfl[nt], acc[mt][nt]);
      }
  }

  // C/D layout: VGPR r -> row (lane<16 ? r : 8+r), col = lane&15
#pragma unroll
  for (int nt = 0; nt < 2; nt++) {
    int col = Nbase + nt * 16 + (lane & 15);
    float bias = b_ih[col] + b_hh[col];
#pragma unroll
    for (int mt = 0; mt < 4; mt++) {
      long row0 = Mbase + mt * 16 + (lane >> 4) * 8;
#pragma unroll
      for (int r = 0; r < 8; r++)
        Xp[(row0 + r) * (long)N + col] = acc[mt][nt][r] + bias;
    }
  }
}

// ---------------------------------------------------------------------------
// Phase 2: persistent reverse scan. 16 WGs x 256 threads.
// WG w owns h-columns [w*32, w*32+32) AND the 4 matching gate column groups,
// so gate exchange is WG-local (LDS). New h crosses WGs through a
// double-buffered split-f16 buffer guarded by an atomic counter barrier.
// ---------------------------------------------------------------------------
#define NWG 16

__global__ __launch_bounds__(256) void recurrent_kernel(
    const float* __restrict__ Xp,          // [T,B,4H] pre-activations (+bias)
    const float* __restrict__ c0,          // [B,H]
    const _Float16* __restrict__ Whh_h, const _Float16* __restrict__ Whh_l,
    _Float16* __restrict__ hb_h0, _Float16* __restrict__ hb_l0,  // buf 0 (holds h0)
    _Float16* __restrict__ hb_h1, _Float16* __restrict__ hb_l1,  // buf 1
    float* __restrict__ out,               // outputs[T,B,H] ++ h_f ++ c_f
    unsigned int* __restrict__ syncp) {
  const int B = 64, H = 512, G = 2048, T = 1024;
  int w    = blockIdx.x;
  int tid  = threadIdx.x;
  int lane = tid & 31;
  int wave = tid >> 5;

  __shared__ float gl[64 * 128];  // raw recurrent-gate slice [B x 128 cols]

  // phase-2 element ownership (fixed across all steps -> c lives in registers)
  int j     = tid & 31;
  int colp2 = w * 32 + j;
  int mrow0 = tid >> 5;  // rows mrow0 + 8k, k=0..7
  float creg[8], hreg[8];
#pragma unroll
  for (int k = 0; k < 8; k++) creg[k] = c0[(mrow0 + k * 8) * H + colp2];

  // phase-1 wave mapping: gate group g = wave>>1, N-tile base
  int grp     = wave >> 1;
  int colbase = grp * 512 + w * 32 + (wave & 1) * 16;

  for (int s = 0; s < T; s++) {
    int t = T - 1 - s;
    const _Float16* rh = (s & 1) ? hb_h1 : hb_h0;
    const _Float16* rl = (s & 1) ? hb_l1 : hb_l0;
    _Float16* wh = (s & 1) ? hb_h0 : hb_h1;
    _Float16* wl = (s & 1) ? hb_l0 : hb_l1;

    // ---- phase 1: gates_slice = h @ W_hh^T for our 16 columns, all 64 rows
    v8f acc[4] = {};
    for (int kb = 0; kb < H; kb += 32) {
      v16h bh = load_fragB(Whh_h + (long)colbase * H + kb, H, lane);
      v16h bl = load_fragB(Whh_l + (long)colbase * H + kb, H, lane);
#pragma unroll
      for (int mt = 0; mt < 4; mt++) {
        v16h ah = load_fragA(rh + (mt * 16) * H + kb, H, lane);
        v16h al = load_fragA(rl + (mt * 16) * H + kb, H, lane);
        acc[mt] = wmma_f16(ah, bh, acc[mt]);
        acc[mt] = wmma_f16(al, bh, acc[mt]);
        acc[mt] = wmma_f16(ah, bl, acc[mt]);
      }
    }
    int lc = wave * 16 + (lane & 15);  // local gate column 0..127
#pragma unroll
    for (int mt = 0; mt < 4; mt++) {
      int m0 = mt * 16 + (lane >> 4) * 8;
#pragma unroll
      for (int r = 0; r < 8; r++) gl[(m0 + r) * 128 + lc] = acc[mt][r];
    }
    __syncthreads();

    // ---- phase 2: gate nonlinearity + state update for our 32 columns
    const float* xp = Xp + (long)t * B * G;
#pragma unroll
    for (int k = 0; k < 8; k++) {
      int m = mrow0 + k * 8;
      float gi = sigmoidf(gl[m * 128 + j]      + xp[m * G +        w * 32 + j]);
      float gf = sigmoidf(gl[m * 128 + 32 + j] + xp[m * G +  512 + w * 32 + j]);
      float gz = sigmoidf(gl[m * 128 + 64 + j] + xp[m * G + 1024 + w * 32 + j]);
      float go = sigmoidf(gl[m * 128 + 96 + j] + xp[m * G + 1536 + w * 32 + j]);
      float cn = gf * creg[k] + gz - gi;
      float hn = sigmoidf(cn) - go;
      creg[k] = cn;
      hreg[k] = hn;
      out[(long)t * B * H + m * H + colp2] = hn;
      _Float16 hh = (_Float16)hn;
      wh[m * H + colp2] = hh;
      wl[m * H + colp2] = (_Float16)(hn - (float)hh);
    }
    __syncthreads();

    // ---- device-wide barrier: release writes, arrive, spin, acquire
    __threadfence();
    if (tid == 0) {
      atomicAdd(syncp, 1u);
      unsigned target = (unsigned)(s + 1) * NWG;
      while (__hip_atomic_load(syncp, __ATOMIC_RELAXED,
                               __HIP_MEMORY_SCOPE_AGENT) < target) {
        __builtin_amdgcn_s_sleep(1);
      }
    }
    __syncthreads();
    __threadfence();
  }

  // finals: h_f = h at t=0 (last computed), c_f likewise
  float* hf = out + (long)T * B * H;
  float* cf = hf + B * H;
#pragma unroll
  for (int k = 0; k < 8; k++) {
    int m = mrow0 + k * 8;
    hf[m * H + colp2] = hreg[k];
    cf[m * H + colp2] = creg[k];
  }
}

// ---------------------------------------------------------------------------
extern "C" void kernel_launch(void* const* d_in, const int* in_sizes, int n_in,
                              void* d_out, int out_size, void* d_ws, size_t ws_size,
                              hipStream_t stream) {
  (void)in_sizes; (void)n_in; (void)out_size; (void)ws_size;
  const int T = 1024, B = 64, I = 512, H = 512;
  const long M = (long)T * B;        // 65536
  const int  G = 4 * H;              // 2048

  const float* input = (const float*)d_in[0];
  const float* h0    = (const float*)d_in[1];
  const float* c0    = (const float*)d_in[2];
  const float* W_ih  = (const float*)d_in[3];
  const float* W_hh  = (const float*)d_in[4];
  const float* b_ih  = (const float*)d_in[5];
  const float* b_hh  = (const float*)d_in[6];
  float* out = (float*)d_out;

  char* ws = (char*)d_ws;
  size_t off = 0;
  float*    Xp    = (float*)(ws + off);    off += (size_t)M * G * 4;      // 512 MB
  _Float16* Ah    = (_Float16*)(ws + off); off += (size_t)M * I * 2;      // 64 MB
  _Float16* Al    = (_Float16*)(ws + off); off += (size_t)M * I * 2;      // 64 MB
  _Float16* Wih_h = (_Float16*)(ws + off); off += (size_t)G * I * 2;      // 2 MB
  _Float16* Wih_l = (_Float16*)(ws + off); off += (size_t)G * I * 2;
  _Float16* Whh_h = (_Float16*)(ws + off); off += (size_t)G * H * 2;
  _Float16* Whh_l = (_Float16*)(ws + off); off += (size_t)G * H * 2;
  _Float16* hb_h0 = (_Float16*)(ws + off); off += (size_t)B * H * 2;
  _Float16* hb_l0 = (_Float16*)(ws + off); off += (size_t)B * H * 2;
  _Float16* hb_h1 = (_Float16*)(ws + off); off += (size_t)B * H * 2;
  _Float16* hb_l1 = (_Float16*)(ws + off); off += (size_t)B * H * 2;
  unsigned* syncp = (unsigned*)(ws + off); off += 256;

  hipMemsetAsync(syncp, 0, sizeof(unsigned), stream);

  // operand splits
  split_f32_kernel<<<4096, 256, 0, stream>>>(input, Ah, Al, (int)(M * I));
  split_f32_kernel<<<1024, 256, 0, stream>>>(W_ih, Wih_h, Wih_l, G * I);
  split_f32_kernel<<<1024, 256, 0, stream>>>(W_hh, Whh_h, Whh_l, G * H);
  split_f32_kernel<<<32,   256, 0, stream>>>(h0, hb_h0, hb_l0, B * H);

  // phase 1: big input-projection GEMM (WMMA)
  dim3 g1(G / 128, (unsigned)(M / 128));
  xproj_gemm_kernel<<<g1, 256, 0, stream>>>(Ah, Al, Wih_h, Wih_l, b_ih, b_hh, Xp);

  // phase 2: persistent reverse scan (WMMA recurrent GEMM per step)
  recurrent_kernel<<<NWG, 256, 0, stream>>>(Xp, c0, Whh_h, Whh_l,
                                            hb_h0, hb_l0, hb_h1, hb_l1,
                                            out, syncp);
}
